// AttnBlock_88484916232484
// MI455X (gfx1250) — matrix operand order repaired and checked by reading, not compiled, not run on hardware
//
#include <hip/hip_runtime.h>

// ---------------------------------------------------------------------------
// AttnBlock for MI455X (gfx1250): C=128, N=16*24*24=9216, batch=1.
// Matrix-op bound (~45 GFLOP vs ~25MB HBM traffic) -> everything on
// v_wmma_f32_16x16x32_bf16. Flash-attention style: no N x N matrix in HBM.
// K/V tiles staged in LDS via CDNA5 async global->LDS copies (ASYNCcnt)
// when the toolchain exposes the builtin; VGPR-staging fallback otherwise.
// Workspace use: 4 * 9216*128 bf16 buffers = 9.4 MB.
// ---------------------------------------------------------------------------

typedef __bf16 bf16;
typedef __attribute__((ext_vector_type(16))) __bf16 v16bf;
typedef __attribute__((ext_vector_type(8)))  __bf16 v8bf;
typedef __attribute__((ext_vector_type(8)))  float  v8f;
// Exact pointee type expected by the async-LDS builtin (from hipcc diagnostic)
typedef int v4i_vs __attribute__((vector_size(4 * sizeof(int))));

#define NTOK 9216   // Z*H*W
#define CDIM 128    // channels
#define BK   64     // keys per flash-attention iteration

#define AS1 __attribute__((address_space(1)))
#define AS3 __attribute__((address_space(3)))

#if __has_builtin(__builtin_amdgcn_global_load_async_to_lds_b128) && \
    __has_builtin(__builtin_amdgcn_s_wait_asynccnt)
#define USE_ASYNC_LDS 1
#else
#define USE_ASYNC_LDS 0
#endif

static __device__ inline v8f wmma_bf16(v16bf a, v16bf b, v8f c) {
  // D = A(16x32 bf16) * B(32x16 bf16) + C(16x16 f32)
  return __builtin_amdgcn_wmma_f32_16x16x32_bf16(
      /*neg_a=*/false, a, /*neg_b=*/false, b,
      /*c_mod=*/(short)0, c, /*reuse_a=*/false, /*reuse_b=*/false);
}

// A-matrix fragment (16x32, bf16), row-major source with contiguous K.
// Lanes 0-15 hold K = 0..15, lanes 16-31 hold
// K = 16..31; p points at (row, k_base) of this lane's row.
static __device__ inline v16bf load_a_frag_row(const bf16* p, int laneHalf) {
  const int lo = laneHalf ? 16 : 0;
  union { v16bf v; v8bf h[2]; } u;
  u.h[0] = *(const v8bf*)(p + lo);
  u.h[1] = *(const v8bf*)(p + lo + 8);
  return u.v;
}

// B-matrix fragment (32x16, bf16): lane holds one column, 16 contiguous K
// values (caller already applied the +16 offset for lanes 16-31).
static __device__ inline v16bf load_b_frag_contig(const bf16* p) {
  return *(const v16bf*)p;   // 32B aligned by construction
}

#if USE_ASYNC_LDS
// 16B direct global->LDS async copy (per lane), tracked by ASYNCcnt.
static __device__ inline void async_copy_b128(const bf16* gsrc, bf16* lds) {
  __builtin_amdgcn_global_load_async_to_lds_b128(
      (AS1 v4i_vs*)gsrc, (AS3 v4i_vs*)lds, /*offset=*/0, /*cpol=*/0);
}
#endif

// ---------------------------------------------------------------------------
// Kernel 1: q = Wq*y + bq ; k = Wk*x + bk ; v = Wv*x + bv   (1x1x1 convs)
// One wave per 16x16 output tile. Outputs (bf16):
//   qw[n][c], kw[m][c]  (token-major, channel-contiguous)
//   vw[c][m]            (channel-major, token-contiguous)
// ---------------------------------------------------------------------------
__global__ __launch_bounds__(256) void qkv_proj_kernel(
    const float* __restrict__ x,  const float* __restrict__ y,
    const float* __restrict__ Wq, const float* __restrict__ bq,
    const float* __restrict__ Wk, const float* __restrict__ bk,
    const float* __restrict__ Wv, const float* __restrict__ bv,
    bf16* __restrict__ qw, bf16* __restrict__ kw, bf16* __restrict__ vw)
{
  const int lane  = threadIdx.x & 31;
  const int wave  = threadIdx.x >> 5;
  const int tile  = blockIdx.x * 8 + wave;          // 0 .. 13823
  const int which = tile / (8 * (NTOK / 16));       // 0=q, 1=k, 2=v
  const int rem   = tile % (8 * (NTOK / 16));
  const int ot    = rem / (NTOK / 16);              // out-channel tile (0..7)
  const int nt    = rem % (NTOK / 16);              // token tile (0..575)

  const float *W, *bias, *in;
  if (which == 0)      { W = Wq; bias = bq; in = y; }
  else if (which == 1) { W = Wk; bias = bk; in = x; }
  else                 { W = Wv; bias = bv; in = x; }

  const int half  = lane >> 4;
  const int l16   = lane & 15;
  const int obase = ot * 16;
  const int nbase = nt * 16;

  v8f acc = {};
  #pragma unroll
  for (int kc = 0; kc < 4; ++kc) {          // C=128 -> 4 x K=32 steps
    // A fragment: weight row (fp32 -> bf16 on the fly)
    const float* wrow = W + (obase + l16) * CDIM + kc * 32;
    const int lo = half ? 16 : 0;
    v16bf af, bfr;
    #pragma unroll
    for (int i = 0; i < 8; ++i) af[i]     = (bf16)wrow[lo + i];
    #pragma unroll
    for (int i = 0; i < 8; ++i) af[8 + i] = (bf16)wrow[lo + 8 + i];
    // B fragment: input column (in is [c][n] fp32, stride NTOK)
    const float* bcol = in + (kc * 32 + half * 16) * NTOK + nbase + l16;
    #pragma unroll
    for (int i = 0; i < 16; ++i) bfr[i] = (bf16)bcol[i * NTOK];
    acc = wmma_bf16(af, bfr, acc);
  }

  // D layout: VGPR r, lanes 0-15 -> row r, lanes 16-31 -> row r+8; col = l16
  #pragma unroll
  for (int r = 0; r < 8; ++r) {
    const int o = obase + r + 8 * half;
    const int n = nbase + l16;
    const float val = acc[r] + bias[o];
    if (which == 2)      vw[o * NTOK + n] = (bf16)val;  // [c][m]
    else if (which == 1) kw[n * CDIM + o] = (bf16)val;  // [m][c]
    else                 qw[n * CDIM + o] = (bf16)val;  // [n][c]
  }
}

// ---------------------------------------------------------------------------
// Kernel 2: flash attention. Each wave owns 16 query rows; the 8 waves of a
// workgroup share K/V tiles of 64 keys staged in LDS (async global->LDS
// copies on CDNA5). Online softmax with wave32 __shfl_xor row reductions;
// P goes through per-wave LDS scratch to convert f32 C-layout -> bf16
// A-layout. Epilogue adds the first +x residual, stores bf16 [n][c].
// ---------------------------------------------------------------------------
__global__ __launch_bounds__(256) void flash_attn_kernel(
    const bf16* __restrict__ qw, const bf16* __restrict__ kw,
    const bf16* __restrict__ vw, const float* __restrict__ x,
    bf16* __restrict__ ow)
{
  __shared__ bf16 Kt[BK][CDIM];      // [key][c]      16 KB
  __shared__ bf16 Vt[CDIM][BK];      // [c][key]      16 KB
  __shared__ bf16 Pt[8][16][BK];     // per-wave P    16 KB

  const int lane  = threadIdx.x & 31;
  const int wave  = threadIdx.x >> 5;
  const int half  = lane >> 4;
  const int l16   = lane & 15;
  const int qbase = (blockIdx.x * 8 + wave) * 16;   // 576 q-tiles / 8 waves

  // Preload Q fragments (4 x 16x32 over the channel dim)
  v16bf aq[4];
  {
    const bf16* qrow = qw + (qbase + l16) * CDIM;
    #pragma unroll
    for (int kc = 0; kc < 4; ++kc)
      aq[kc] = load_a_frag_row(qrow + kc * 32, half);
  }

  v8f zero = {};
  v8f o[8];                                  // 16 x 128 f32 accumulator
  #pragma unroll
  for (int t = 0; t < 8; ++t) o[t] = zero;
  float mrun[8], lrun[8];
  #pragma unroll
  for (int r = 0; r < 8; ++r) { mrun[r] = -3.0e38f; lrun[r] = 0.0f; }

  // softmax scale folded with log2(e):  exp(s/sqrt(C)) = exp2(s*sc)
  const float sc = 0.08838834764831845f * 1.4426950408889634f;

  for (int kb = 0; kb < NTOK / BK; ++kb) {
    const int kbase = kb * BK;
    __syncthreads();                         // previous tile fully consumed

    // Cooperative LDS staging: Kt = 1024 x 16B, Vt = 1024 x 16B; 256 threads.
#if USE_ASYNC_LDS
    #pragma unroll
    for (int i = 0; i < 4; ++i) {            // Kt (contiguous rows)
      const int idx = threadIdx.x + 256 * i; // 0..1023
      async_copy_b128(kw + kbase * CDIM + idx * 8, &Kt[0][0] + idx * 8);
    }
    #pragma unroll
    for (int i = 0; i < 4; ++i) {            // Vt (row-strided source)
      const int idx = threadIdx.x + 256 * i;
      const int c   = idx >> 3;              // 0..127
      const int ch  = idx & 7;               // 16B chunk within row
      async_copy_b128(vw + c * NTOK + kbase + ch * 8, &Vt[c][0] + ch * 8);
    }
    __builtin_amdgcn_s_wait_asynccnt(0);     // this wave's copies landed
#else
    {
      uint4*       dstK = (uint4*)&Kt[0][0];
      const uint4* srcK = (const uint4*)(kw + kbase * CDIM);
      #pragma unroll
      for (int i = 0; i < 4; ++i)
        dstK[threadIdx.x + 256 * i] = srcK[threadIdx.x + 256 * i];
      #pragma unroll
      for (int i = 0; i < 4; ++i) {
        const int idx = threadIdx.x + 256 * i;
        const int c   = idx >> 3;
        const int ch  = idx & 7;
        ((uint4*)&Vt[c][0])[ch] =
            *(const uint4*)(vw + c * NTOK + kbase + ch * 8);
      }
    }
#endif
    __syncthreads();                         // all waves' tiles visible

    // S = Q^T K : 4 column tiles x 4 channel chunks = 16 WMMAs
    v8f s[4];
    #pragma unroll
    for (int j = 0; j < 4; ++j) {
      v8f a = zero;
      #pragma unroll
      for (int kc = 0; kc < 4; ++kc) {
        const bf16* p = &Kt[j * 16 + l16][kc * 32 + half * 16];
        a = wmma_bf16(aq[kc], load_b_frag_contig(p), a);
      }
      s[j] = a;
    }

    // Online softmax. Row of S = 16 lanes within a half of one VGPR;
    // xor masks 1..8 reduce within each 16-lane group independently.
    float mnew[8], alpha[8];
    #pragma unroll
    for (int r = 0; r < 8; ++r) {
      float mx = fmaxf(fmaxf(s[0][r], s[1][r]), fmaxf(s[2][r], s[3][r]));
      #pragma unroll
      for (int d = 1; d < 16; d <<= 1) mx = fmaxf(mx, __shfl_xor(mx, d, 32));
      mnew[r]  = fmaxf(mrun[r], mx);
      alpha[r] = exp2f((mrun[r] - mnew[r]) * sc);
      mrun[r]  = mnew[r];
    }
    #pragma unroll
    for (int r = 0; r < 8; ++r) {
      float rs = 0.0f;
      #pragma unroll
      for (int j = 0; j < 4; ++j) {
        const float p = exp2f((s[j][r] - mnew[r]) * sc);
        rs += p;
        Pt[wave][r + 8 * half][j * 16 + l16] = (bf16)p;   // C-layout -> LDS
      }
      #pragma unroll
      for (int d = 1; d < 16; d <<= 1) rs += __shfl_xor(rs, d, 32);
      lrun[r] = lrun[r] * alpha[r] + rs;
      #pragma unroll
      for (int t = 0; t < 8; ++t) o[t][r] = o[t][r] * alpha[r];
    }

    // O += P V : 2 key chunks x 8 channel tiles = 16 WMMAs
    #pragma unroll
    for (int k2 = 0; k2 < 2; ++k2) {
      const v16bf ap = load_a_frag_row(&Pt[wave][l16][k2 * 32], half);
      #pragma unroll
      for (int t = 0; t < 8; ++t) {
        const bf16* p = &Vt[t * 16 + l16][k2 * 32 + half * 16];
        o[t] = wmma_bf16(ap, load_b_frag_contig(p), o[t]);
      }
    }
  }

  // Normalize, add first residual (+x), store bf16 [n][c]
  #pragma unroll
  for (int t = 0; t < 8; ++t) {
    #pragma unroll
    for (int r = 0; r < 8; ++r) {
      const int n = qbase + r + 8 * half;
      const int c = t * 16 + l16;
      const float val = o[t][r] / lrun[r] + x[c * NTOK + n];
      ow[n * CDIM + c] = (bf16)val;
    }
  }
}

// ---------------------------------------------------------------------------
// Kernel 3: out = Wp * (o + x) + bp + x   (fp32 output, [c][n] order)
// ---------------------------------------------------------------------------
__global__ __launch_bounds__(256) void proj_out_kernel(
    const bf16* __restrict__ ow, const float* __restrict__ Wp,
    const float* __restrict__ bp, const float* __restrict__ x,
    float* __restrict__ out)
{
  const int lane = threadIdx.x & 31;
  const int wave = threadIdx.x >> 5;
  const int tile = blockIdx.x * 8 + wave;         // 8 * 576 tiles
  const int ot   = tile / (NTOK / 16);
  const int nt   = tile % (NTOK / 16);
  const int half = lane >> 4, l16 = lane & 15;
  const int obase = ot * 16, nbase = nt * 16;

  v8f acc = {};
  #pragma unroll
  for (int kc = 0; kc < 4; ++kc) {
    const float* wrow = Wp + (obase + l16) * CDIM + kc * 32;
    const int lo = half ? 16 : 0;
    v16bf af;
    #pragma unroll
    for (int i = 0; i < 8; ++i) af[i]     = (bf16)wrow[lo + i];
    #pragma unroll
    for (int i = 0; i < 8; ++i) af[8 + i] = (bf16)wrow[lo + 8 + i];
    // B fragment: ow[n][c] is channel-contiguous -> one 32B load
    const bf16* p = ow + (nbase + l16) * CDIM + kc * 32 + half * 16;
    acc = wmma_bf16(af, load_b_frag_contig(p), acc);
  }
  #pragma unroll
  for (int r = 0; r < 8; ++r) {
    const int o = obase + r + 8 * half;
    const int n = nbase + l16;
    out[o * NTOK + n] = acc[r] + bp[o] + x[o * NTOK + n];
  }
}

// ---------------------------------------------------------------------------
extern "C" void kernel_launch(void* const* d_in, const int* in_sizes, int n_in,
                              void* d_out, int out_size, void* d_ws, size_t ws_size,
                              hipStream_t stream) {
  (void)in_sizes; (void)n_in; (void)out_size; (void)ws_size;
  const float* x  = (const float*)d_in[0];
  const float* y  = (const float*)d_in[1];
  const float* Wq = (const float*)d_in[2];
  const float* bq = (const float*)d_in[3];
  const float* Wk = (const float*)d_in[4];
  const float* bk = (const float*)d_in[5];
  const float* Wv = (const float*)d_in[6];
  const float* bv = (const float*)d_in[7];
  const float* Wp = (const float*)d_in[8];
  const float* bp = (const float*)d_in[9];
  float* out = (float*)d_out;

  char* ws = (char*)d_ws;
  const size_t SZ = (size_t)NTOK * CDIM * sizeof(bf16);   // 2.36 MB each
  bf16* qw = (bf16*)(ws);
  bf16* kw = (bf16*)(ws + SZ);
  bf16* vw = (bf16*)(ws + 2 * SZ);
  bf16* ow = (bf16*)(ws + 3 * SZ);

  // 3 x (8 o-tiles) x (576 n-tiles) waves, 8 waves/block
  qkv_proj_kernel<<<dim3(1728), dim3(256), 0, stream>>>(
      x, y, Wq, bq, Wk, bk, Wv, bv, qw, kw, vw);
  // 576 q-tiles / 8 waves per block
  flash_attn_kernel<<<dim3(72), dim3(256), 0, stream>>>(qw, kw, vw, x, ow);
  // (8 o-tiles) x (576 n-tiles) / 8 waves per block
  proj_out_kernel<<<dim3(576), dim3(256), 0, stream>>>(ow, Wp, bp, x, out);
}
